// LSTM_58025008169337
// MI455X (gfx1250) — matrix operand order repaired and checked by reading, not compiled, not run on hardware
//
#include <hip/hip_runtime.h>
#include <hip/hip_bf16.h>
#include <stdint.h>

// Problem dims (fixed by reference)
#define B_  256
#define T_  512
#define D_  256
#define H_  1024
#define C_  10
#define G4  (4 * H_)     // 4096 concatenated gate columns

typedef __attribute__((ext_vector_type(16))) __bf16 v16bf;
typedef __attribute__((ext_vector_type(8)))  float  v8f;

__device__ __forceinline__ uint16_t f32_to_bf16_rne(float f) {
    uint32_t u = __float_as_uint(f);
    u += 0x7FFFu + ((u >> 16) & 1u);        // round-to-nearest-even
    return (uint16_t)(u >> 16);
}
__device__ __forceinline__ float bf16u_to_f32(uint16_t h) {
    return __uint_as_float(((uint32_t)h) << 16);
}

// ---------------------------------------------------------------------------
// Kernel 1: P[v][gate*H + j] = emb[v,:] . W{gate}x[:,j] + b{gate}[j]
// VOCAB=3 -> xproj collapses to a 3x4096 lookup table (removes 275 GFLOP GEMM)
// ---------------------------------------------------------------------------
__global__ void k_proj_P(const float* __restrict__ emb,
                         const float* __restrict__ Wgx, const float* __restrict__ bg,
                         const float* __restrict__ Wix, const float* __restrict__ bi,
                         const float* __restrict__ Wfx, const float* __restrict__ bf_,
                         const float* __restrict__ Wox, const float* __restrict__ bo,
                         float* __restrict__ P) {
    int i = blockIdx.x * blockDim.x + threadIdx.x;
    if (i >= 3 * G4) return;
    int v    = i / G4;
    int col  = i % G4;
    int gate = col / H_;
    int j    = col % H_;
    const float* Wx; const float* b;
    switch (gate) {
        case 0:  Wx = Wgx; b = bg;  break;
        case 1:  Wx = Wix; b = bi;  break;
        case 2:  Wx = Wfx; b = bf_; break;
        default: Wx = Wox; b = bo;  break;
    }
    float acc = b[j];
    for (int d = 0; d < D_; ++d) acc += emb[v * D_ + d] * Wx[d * H_ + j];
    P[i] = acc;
}

// ---------------------------------------------------------------------------
// Kernel 2: swizzle Wh (4x 1024x1024 f32) into fragment-major bf16.
// One thread per (gate, ntile, kb, lane): emits that lane's contiguous
// 16-element (32B) WMMA B-fragment -> perfectly coalesced 1KB/wave writes.
// Layout: Wsw[(((gate*64 + ntile)*32 + kb)*32 + lane)*16 + e]
//   = Wh_gate[kb*32 + (lane>>4)*16 + e][ntile*16 + (lane&15)]
// ---------------------------------------------------------------------------
__global__ void k_swizzle_Wh(const float* __restrict__ Wgh, const float* __restrict__ Wih,
                             const float* __restrict__ Wfh, const float* __restrict__ Woh,
                             uint16_t* __restrict__ Wsw) {
    int i = blockIdx.x * blockDim.x + threadIdx.x;   // 0 .. 4*64*32*32 = 262144
    int lane =  i        & 31;
    int kb   = (i >> 5)  & 31;
    int nt   = (i >> 10) & 63;
    int gate =  i >> 16;
    const float* W = (gate == 0) ? Wgh : (gate == 1) ? Wih : (gate == 2) ? Wfh : Woh;
    int kbase = kb * 32 + (lane >> 4) * 16;
    int j     = nt * 16 + (lane & 15);
    alignas(16) uint16_t frag[16];
#pragma unroll
    for (int e = 0; e < 16; ++e)
        frag[e] = f32_to_bf16_rne(W[(size_t)(kbase + e) * H_ + j]);
    uint16_t* dst = Wsw + ((size_t)i << 4);
    *(uint4*)dst       = *(const uint4*)frag;
    *(uint4*)(dst + 8) = *(const uint4*)(frag + 8);
}

// Kernel 3: zero initial h and c state
__global__ void k_init(float* __restrict__ cst, uint16_t* __restrict__ h0) {
    int i = blockIdx.x * blockDim.x + threadIdx.x;
    if (i < B_ * H_) { cst[i] = 0.f; h0[i] = 0; }
}

// ---------------------------------------------------------------------------
// Kernel 4: one LSTM timestep.
// Grid: 128 blocks = (B/32) x (H/64). Block: 128 threads = 4 waves.
// Wave w owns gate w's 32(B) x 64(j) tile: 8 accumulator tiles; K=1024 done
// as 2 LDS-staged chunks of 512, 16 steps of v_wmma_f32_16x16x32_bf16 each
// reusing one B fragment for two M-tiles (8 wmma : 8 b128 B-loads per kb).
// Cross-gate cell update fused via LDS (zbuf aliases the h-tile buffer).
// ---------------------------------------------------------------------------
#define KC   512                 // K chunk staged in LDS
#define HTS  (KC + 8)            // padded row stride (halfwords)

__global__ __launch_bounds__(128) void k_lstm_step(
    const uint16_t* __restrict__ hin,   // [B][H] bf16
    uint16_t*       __restrict__ hout,  // [B][H] bf16
    float*          __restrict__ cst,   // [B][H] f32
    const uint16_t* __restrict__ Wsw,   // swizzled Wh, bf16
    const float*    __restrict__ P,     // [3][4096] xproj+bias table
    const int*      __restrict__ x,     // [B][T] int32
    int t)
{
    __shared__ __align__(16) char smem[32 * HTS * 2];   // 33,280 B
    uint16_t (*hTile)[HTS]   = (uint16_t(*)[HTS])smem;  // 32 x 512 bf16 h chunk
    float    (*zbuf)[32][64] = (float(*)[32][64])smem;  // aliased: 4x32x64 f32
    __shared__ int xv[32];

    const int tid  = threadIdx.x;
    const int lane = tid & 31;
    const int gate = tid >> 5;               // wave id == gate id
    const int bblk = blockIdx.x & 7;
    const int jblk = blockIdx.x >> 3;
    const int b0   = bblk * 32;
    const int j0   = jblk * 64;

    if (tid < 32) xv[tid] = x[(size_t)(b0 + tid) * T_ + t];

    v8f zero = {0.f, 0.f, 0.f, 0.f, 0.f, 0.f, 0.f, 0.f};
    v8f acc[2][4];
#pragma unroll
    for (int m = 0; m < 2; ++m)
#pragma unroll
        for (int n = 0; n < 4; ++n) acc[m][n] = zero;

    // Base of this (gate, jblk) strip in the swizzled weights.
    const uint16_t* wg = Wsw + ((size_t)(gate * 64 + jblk * 4) << 14);

    const int arow = lane & 15;
    for (int kc = 0; kc < 2; ++kc) {
        __syncthreads();   // previous chunk fully consumed (also covers xv)
        // Stage 32 x 512 bf16 h chunk: 2048 16B pieces, 16 per thread.
        for (int chunk = tid; chunk < 32 * (KC / 8); chunk += 128) {
            int row = chunk >> 6;            // KC/8 = 64 chunks per row
            int cc  = chunk & 63;
            uint4 v = *(const uint4*)(hin + (size_t)(b0 + row) * H_ + kc * KC + cc * 8);
            *(uint4*)(&hTile[row][cc * 8]) = v;
        }
        __syncthreads();

        for (int kb8 = 0; kb8 < KC / 32; ++kb8) {
            const int kb = kc * (KC / 32) + kb8;   // global 32-wide K block
            // A fragments (contiguous k order, same as the Wsw B packing):
            // lanes 0-15: K k0+0..15 ; lanes 16-31: K k0+16..31.
            int koff = kb8 * 32 + (lane >> 4) * 16;
            v16bf a0, a1;
            ((uint4*)&a0)[0] = *(const uint4*)(&hTile[arow][koff]);
            ((uint4*)&a0)[1] = *(const uint4*)(&hTile[arow][koff + 8]);
            ((uint4*)&a1)[0] = *(const uint4*)(&hTile[16 + arow][koff]);
            ((uint4*)&a1)[1] = *(const uint4*)(&hTile[16 + arow][koff + 8]);

#pragma unroll
            for (int n = 0; n < 4; ++n) {
                const uint16_t* bp = wg + ((size_t)n << 14) + (size_t)kb * 512 + lane * 16;
                v16bf bm;
                ((uint4*)&bm)[0] = *(const uint4*)(bp);
                ((uint4*)&bm)[1] = *(const uint4*)(bp + 8);
                acc[0][n] = __builtin_amdgcn_wmma_f32_16x16x32_bf16(
                                false, a0, false, bm, (short)0, acc[0][n], false, false);
                acc[1][n] = __builtin_amdgcn_wmma_f32_16x16x32_bf16(
                                false, a1, false, bm, (short)0, acc[1][n], false, false);
            }
        }
    }

    __syncthreads();   // all waves done reading hTile -> safe to alias as zbuf

    // Epilogue: z = tanh(acc + xproj_gathered); share across gates via LDS.
    // C/D layout: VGPR r -> M = r (lanes 0-15) / r+8 (lanes 16-31), N = lane&15.
#pragma unroll
    for (int m = 0; m < 2; ++m) {
#pragma unroll
        for (int n = 0; n < 4; ++n) {
#pragma unroll
            for (int r = 0; r < 8; ++r) {
                int row = m * 16 + r + ((lane >> 4) << 3);
                int col = n * 16 + (lane & 15);
                float xp = P[(size_t)xv[row] * G4 + gate * H_ + j0 + col];
                zbuf[gate][row][col] = tanhf(acc[m][n][r] + xp);
            }
        }
    }
    __syncthreads();

    // Fused cell update: c = g*i + c*f ; h = tanh(c)*o
    {
        int row = tid >> 2;              // 0..31
        int cg  = (tid & 3) * 16;        // 16-col group
#pragma unroll
        for (int q = 0; q < 16; ++q) {
            int col = cg + q;
            float g  = zbuf[0][row][col];
            float ii = zbuf[1][row][col];
            float f  = zbuf[2][row][col];
            float o  = zbuf[3][row][col];
            size_t idx = (size_t)(b0 + row) * H_ + j0 + col;
            float c_new = g * ii + cst[idx] * f;
            cst[idx]  = c_new;
            hout[idx] = f32_to_bf16_rne(tanhf(c_new) * o);
        }
    }
}

// ---------------------------------------------------------------------------
// Kernel 5: head p = h @ W_ph + b_p, then log_softmax (C=10, trivial VALU)
// ---------------------------------------------------------------------------
__global__ void k_head(const uint16_t* __restrict__ h,
                       const float* __restrict__ Wph, const float* __restrict__ bp,
                       float* __restrict__ out) {
    int b = blockIdx.x * blockDim.x + threadIdx.x;
    if (b >= B_) return;
    float p[C_];
#pragma unroll
    for (int c = 0; c < C_; ++c) p[c] = bp[c];
    for (int k = 0; k < H_; ++k) {
        float hv = bf16u_to_f32(h[(size_t)b * H_ + k]);
#pragma unroll
        for (int c = 0; c < C_; ++c) p[c] += hv * Wph[k * C_ + c];
    }
    float m = p[0];
#pragma unroll
    for (int c = 1; c < C_; ++c) m = fmaxf(m, p[c]);
    float s = 0.f;
#pragma unroll
    for (int c = 0; c < C_; ++c) s += __expf(p[c] - m);
    float lse = m + __logf(s);
#pragma unroll
    for (int c = 0; c < C_; ++c) out[b * C_ + c] = p[c] - lse;
}

// ---------------------------------------------------------------------------
extern "C" void kernel_launch(void* const* d_in, const int* in_sizes, int n_in,
                              void* d_out, int out_size, void* d_ws, size_t ws_size,
                              hipStream_t stream) {
    const int*   x   = (const int*)  d_in[0];
    const float* emb = (const float*)d_in[1];
    const float* Wgx = (const float*)d_in[2];
    const float* Wgh = (const float*)d_in[3];
    const float* bg  = (const float*)d_in[4];
    const float* Wix = (const float*)d_in[5];
    const float* Wih = (const float*)d_in[6];
    const float* bi  = (const float*)d_in[7];
    const float* Wfx = (const float*)d_in[8];
    const float* Wfh = (const float*)d_in[9];
    const float* bf  = (const float*)d_in[10];
    const float* Wox = (const float*)d_in[11];
    const float* Woh = (const float*)d_in[12];
    const float* bo  = (const float*)d_in[13];
    const float* Wph = (const float*)d_in[14];
    const float* bp  = (const float*)d_in[15];

    // Workspace carve-up (all offsets well aligned): ~10.5 MB total
    char* ws = (char*)d_ws;
    uint16_t* Wsw = (uint16_t*)ws;  ws += (size_t)4 * 1024 * 1024 * 2;  // 8 MB swizzled Wh (bf16)
    float*    P   = (float*)ws;     ws += (size_t)3 * G4 * 4;           // 48 KB xproj table
    float*    cst = (float*)ws;     ws += (size_t)B_ * H_ * 4;          // 1 MB cell state
    uint16_t* h0  = (uint16_t*)ws;  ws += (size_t)B_ * H_ * 2;          // 512 KB h ping
    uint16_t* h1  = (uint16_t*)ws;                                      // 512 KB h pong

    k_proj_P   <<<(3 * G4 + 255) / 256, 256, 0, stream>>>(emb, Wgx, bg, Wix, bi,
                                                           Wfx, bf, Wox, bo, P);
    k_swizzle_Wh<<<262144 / 256, 256, 0, stream>>>(Wgh, Wih, Wfh, Woh, Wsw);
    k_init     <<<(B_ * H_) / 256, 256, 0, stream>>>(cst, h0);

    uint16_t* hb[2] = {h0, h1};
    for (int t = 0; t < T_; ++t) {
        k_lstm_step<<<128, 128, 0, stream>>>(hb[t & 1], hb[(t + 1) & 1],
                                             cst, Wsw, P, x, t);
    }
    // T even -> final h landed back in h0
    k_head<<<1, 256, 0, stream>>>(h0, Wph, bp, (float*)d_out);
}
